// MultiHeadAttention_62302795596088
// MI455X (gfx1250) — matrix operand order repaired and checked by reading, not compiled, not run on hardware
//
#include <hip/hip_runtime.h>
#include <hip/hip_bf16.h>

// ---------------------------------------------------------------------------
// MHA forward for MI455X (gfx1250, wave32, WMMA, TDM).
//   1) x -> bf16; W* -> bf16 transposed [N,K] (so B fragments are contiguous-k
//      direct global loads -> LDS-free GEMMs, weights live in 192MB L2)
//   2) QKV GEMMs: wave tile 32x64, 8 v_wmma_f32_16x16x32_bf16 per K-step.
//      Q,K written [B,H,S,HD]; V written [B,H,HD,S] (head-transposed).
//   3) Flash attention, KV tile = 64 keys, 16 WMMAs/iter; V tile staged into
//      LDS by the Tensor Data Mover (tensor_load_to_lds + s_wait_tensorcnt).
//   4) Output projection GEMM -> f32.
// ---------------------------------------------------------------------------

typedef __attribute__((ext_vector_type(16))) __bf16         v16bf;
typedef __attribute__((ext_vector_type(16))) unsigned short v16u;
typedef __attribute__((ext_vector_type(8)))  unsigned short v8u;
typedef __attribute__((ext_vector_type(8)))  float          v8f;
typedef __attribute__((ext_vector_type(4)))  unsigned int   v4u;
typedef __attribute__((ext_vector_type(8)))  int            v8i;
typedef __attribute__((ext_vector_type(4)))  int            v4i;

constexpr int BB  = 4;
constexpr int SS  = 2048;
constexpr int DD  = 1024;
constexpr int HH  = 16;
constexpr int HDD = 64;
constexpr int MM  = BB * SS;   // 8192 rows

#define WMMA_BF16(a, b, c) \
  __builtin_amdgcn_wmma_f32_16x16x32_bf16(false, (a), false, (b), (short)0, (c), false, false)

#if __has_builtin(__builtin_amdgcn_tensor_load_to_lds)
#define HAVE_TDM 1
#else
#define HAVE_TDM 0
#endif

__device__ __forceinline__ unsigned short f32_to_bf16(float f) {
  unsigned int u = __float_as_uint(f);
  u += 0x7FFFu + ((u >> 16) & 1u);   // round-to-nearest-even
  return (unsigned short)(u >> 16);
}

__device__ __forceinline__ v16bf combine16(v8u lo, v8u hi) {
  v16u r;
#pragma unroll
  for (int i = 0; i < 8; ++i) { r[i] = lo[i]; r[8 + i] = hi[i]; }
  return __builtin_bit_cast(v16bf, r);
}

// ---------------------------------------------------------------------------
// Kernel 1a: elementwise f32 -> bf16
// ---------------------------------------------------------------------------
__global__ __launch_bounds__(256) void cvt_bf16_kernel(
    const float* __restrict__ src, unsigned short* __restrict__ dst, int n) {
  int i = blockIdx.x * blockDim.x + threadIdx.x;
  if (i < n) dst[i] = f32_to_bf16(src[i]);
}

// ---------------------------------------------------------------------------
// Kernel 1b: W [K,N] f32 -> Wt [N,K] bf16 (32x32 LDS-tiled transpose)
// ---------------------------------------------------------------------------
__global__ __launch_bounds__(256) void cvt_wt_kernel(
    const float* __restrict__ W, unsigned short* __restrict__ Wt) {
  __shared__ float tile[32][33];
  const int bx = blockIdx.x * 32;  // k base
  const int by = blockIdx.y * 32;  // n base
  const int tx = threadIdx.x & 31;
  const int ty = threadIdx.x >> 5;  // 0..7
#pragma unroll
  for (int i = 0; i < 32; i += 8)
    tile[ty + i][tx] = W[(size_t)(bx + ty + i) * DD + by + tx];
  __syncthreads();
#pragma unroll
  for (int i = 0; i < 32; i += 8)
    Wt[(size_t)(by + ty + i) * DD + bx + tx] = f32_to_bf16(tile[tx][ty + i]);
}

// ---------------------------------------------------------------------------
// Kernel 2: LDS-free GEMM  C = A[8192,1024] @ Wt^T + bias
// Block 256 thr = 8 waves as 4 row-groups x 2 col-groups; wave tile 32x64.
// layout 0: scatter bf16 into [B,H,S,HD];  layout 1: into [B,H,HD,S].
// ---------------------------------------------------------------------------
__global__ __launch_bounds__(256) void gemm_qkv_kernel(
    const unsigned short* __restrict__ A,    // [8192,1024] bf16 row-major
    const unsigned short* __restrict__ Wt,   // [N=1024, K=1024] bf16
    const float* __restrict__ bias,
    unsigned short* __restrict__ out, int layout)
{
  const int tid  = threadIdx.x;
  const int wave = tid >> 5;
  const int lane = tid & 31;
  const int half = lane >> 4;
  const int l16  = lane & 15;

  const int rowBase = blockIdx.y * 128 + (wave >> 1) * 32;
  const int nBase   = blockIdx.x * 128 + (wave & 1) * 64;

  v8f acc[2][4] = {};

  for (int k0 = 0; k0 < DD; k0 += 32) {
    v16bf afrag[2];
#pragma unroll
    for (int r = 0; r < 2; ++r) {
      const unsigned short* arow = A + (size_t)(rowBase + r * 16 + l16) * DD + k0;
      afrag[r] = combine16(*(const v8u*)(arow + half * 8),
                           *(const v8u*)(arow + 16 + half * 8));
      __builtin_prefetch(arow + 128, 0, 3);   // next K blocks of A
    }
#pragma unroll
    for (int t = 0; t < 4; ++t) {
      const unsigned short* brow =
          Wt + (size_t)(nBase + t * 16 + l16) * DD + k0 + half * 8;
      const v16bf bfrag = combine16(*(const v8u*)brow, *(const v8u*)(brow + 16));
      acc[0][t] = WMMA_BF16(afrag[0], bfrag, acc[0][t]);
      acc[1][t] = WMMA_BF16(afrag[1], bfrag, acc[1][t]);
    }
  }

#pragma unroll
  for (int t = 0; t < 4; ++t) {
    const int gcol = nBase + t * 16 + l16;
    const float bv = bias[gcol];
    const int h  = gcol >> 6;
    const int hd = gcol & (HDD - 1);
#pragma unroll
    for (int r = 0; r < 2; ++r) {
#pragma unroll
      for (int v = 0; v < 8; ++v) {
        const int grow = rowBase + r * 16 + half * 8 + v;  // C layout: M=v+8*half
        const int b = grow >> 11;
        const int s = grow & (SS - 1);
        const unsigned short val = f32_to_bf16(acc[r][t][v] + bv);
        if (layout == 0)
          out[(size_t)(((b * HH + h) * SS) + s) * HDD + hd] = val;    // [B,H,S,HD]
        else
          out[(size_t)(((b * HH + h) * HDD) + hd) * SS + s] = val;    // [B,H,HD,S]
      }
    }
  }
}

// ---------------------------------------------------------------------------
// TDM: stage 64(hd) x 64(key) bf16 tile of head-transposed V into LDS.
// Tensor = [HD rows x S cols] row-major, tile start at column kv.
// ---------------------------------------------------------------------------
#if HAVE_TDM
__device__ __forceinline__ void tdm_load_v_tile(const unsigned short* gtile,
                                                unsigned int ldsAddr) {
  const unsigned long long ga = (unsigned long long)(size_t)gtile;
  v4u g0 = {0u, 0u, 0u, 0u};
  g0[0] = 1u;                                            // count=1, user mode
  g0[1] = ldsAddr;                                       // LDS byte address
  g0[2] = (unsigned int)(ga & 0xFFFFFFFFu);              // global_addr[31:0]
  g0[3] = (unsigned int)((ga >> 32) & 0x01FFFFFFu) | (2u << 30);  // addr hi | type=2
  v8i g1 = {0, 0, 0, 0, 0, 0, 0, 0};
  g1[0] = (int)(1u << 16);        // workgroup_mask=0, data_size=1 (2 bytes)
  g1[1] = (int)((unsigned)SS << 16);   // tensor_dim0[15:0]=S (keys)
  g1[2] = (int)((unsigned)HDD << 16);  // tensor_dim0 hi=0 | tensor_dim1[15:0]=HD
  g1[3] = (int)(64u << 16);            // tensor_dim1 hi=0 | tile_dim0=64 keys
  g1[4] = 64;                          // tile_dim1=64 hd rows, tile_dim2=0
  g1[5] = SS;                          // tensor_dim0_stride = S (row pitch)
  g1[6] = 0;
  g1[7] = 0;
  v4i g2 = {0, 0, 0, 0};
  v4i g3 = {0, 0, 0, 0};
#if defined(__clang_major__) && __clang_major__ >= 23
  v8i g4 = {0, 0, 0, 0, 0, 0, 0, 0};
  __builtin_amdgcn_tensor_load_to_lds(g0, g1, g2, g3, g4, 0);
#else
  __builtin_amdgcn_tensor_load_to_lds(g0, g1, g2, g3, 0);
#endif
}
#endif

// ---------------------------------------------------------------------------
// Kernel 3: flash attention. Block = 128 threads (4 waves) <-> 64 q rows of
// one (b,h). KV tiles of 64 keys: 8 QK^T WMMAs + 8 PV WMMAs per iteration.
// ---------------------------------------------------------------------------
__global__ __launch_bounds__(128) void attn_kernel(
    const unsigned short* __restrict__ Q,    // [B,H,S,HD] bf16
    const unsigned short* __restrict__ K,    // [B,H,S,HD] bf16
    const unsigned short* __restrict__ Vt,   // [B,H,HD,S] bf16
    unsigned short* __restrict__ attnOut)    // [B,S,D] bf16
{
  __shared__ unsigned short ldsV[64 * 64];     // [hd][key]
  __shared__ unsigned short ldsP[4][16 * 64];  // per-wave P tile

  const int tid  = threadIdx.x;
  const int wave = tid >> 5;
  const int lane = tid & 31;
  const int half = lane >> 4;
  const int l16  = lane & 15;

  const int qBase = blockIdx.x * 64;
  const int bh    = blockIdx.y;
  const int b     = bh >> 4;
  const int h     = bh & (HH - 1);
  const int qRowW = qBase + wave * 16;

  const unsigned short* Qb  = Q  + (size_t)((b * HH + h) * SS) * HDD;
  const unsigned short* Kb  = K  + (size_t)((b * HH + h) * SS) * HDD;
  const unsigned short* Vtb = Vt + (size_t)((b * HH + h) * HDD) * SS;

  // Q fragments for the two 32-deep chunks of HD
  v16bf aq[2];
#pragma unroll
  for (int c = 0; c < 2; ++c) {
    const unsigned short* qrow = Qb + (size_t)(qRowW + l16) * HDD + c * 32;
    aq[c] = combine16(*(const v8u*)(qrow + half * 8),
                      *(const v8u*)(qrow + 16 + half * 8));
  }

  v8f o[4] = {};
  float m[8], l[8];
#pragma unroll
  for (int v = 0; v < 8; ++v) { m[v] = -1e30f; l[v] = 0.0f; }

  // softmax in exp2 domain: softmax(x/8) == softmax2(x * 0.125*log2(e))
  const float scale2 = 0.125f * 1.44269504088896340736f;

  for (int kv = 0; kv < qBase + 64; kv += 64) {
    __syncthreads();
#if HAVE_TDM
    if (wave == 0) {
      tdm_load_v_tile(Vtb + kv, (unsigned int)(size_t)(void*)ldsV);
      __builtin_amdgcn_s_wait_tensorcnt(0);
    }
#else
    {  // manual stage: 64 rows(hd) x 64 keys, layouts already match
      const int r  = tid >> 1;
      const int c0 = (tid & 1) * 32;
#pragma unroll
      for (int j = 0; j < 4; ++j)
        *(v8u*)(ldsV + r * 64 + c0 + j * 8) =
            *(const v8u*)(Vtb + (size_t)r * SS + kv + c0 + j * 8);
    }
#endif
    __syncthreads();

    // ---- S = Q K^T : 4 key-subtiles x 2 d-chunks ----
    v8f sacc[4] = {};
#pragma unroll
    for (int sc = 0; sc < 4; ++sc) {
#pragma unroll
      for (int c = 0; c < 2; ++c) {
        const unsigned short* krow =
            Kb + (size_t)(kv + sc * 16 + l16) * HDD + c * 32 + half * 8;
        sacc[sc] = WMMA_BF16(aq[c],
                             combine16(*(const v8u*)krow, *(const v8u*)(krow + 16)),
                             sacc[sc]);
      }
    }

    // ---- scale + causal mask + row max ----
    float tmax[8];
#pragma unroll
    for (int v = 0; v < 8; ++v) {
      const int qRow = qRowW + half * 8 + v;
      float mx = -1e30f;
#pragma unroll
      for (int sc = 0; sc < 4; ++sc) {
        float s = sacc[sc][v] * scale2;
        if (kv + sc * 16 + l16 > qRow) s = -1e30f;
        sacc[sc][v] = s;
        mx = fmaxf(mx, s);
      }
      mx = fmaxf(mx, __shfl_xor(mx, 1, 32));
      mx = fmaxf(mx, __shfl_xor(mx, 2, 32));
      mx = fmaxf(mx, __shfl_xor(mx, 4, 32));
      mx = fmaxf(mx, __shfl_xor(mx, 8, 32));
      tmax[v] = mx;
    }

    // ---- online softmax update; P -> wave-private LDS (bf16) ----
#pragma unroll
    for (int v = 0; v < 8; ++v) {
      const float mnew  = fmaxf(m[v], tmax[v]);
      const float alpha = __builtin_amdgcn_exp2f(m[v] - mnew);
      float ps = 0.0f;
      const int rowL = half * 8 + v;
#pragma unroll
      for (int sc = 0; sc < 4; ++sc) {
        const float p = __builtin_amdgcn_exp2f(sacc[sc][v] - mnew);
        ps += p;
        ldsP[wave][rowL * 64 + sc * 16 + l16] = f32_to_bf16(p);
      }
      ps += __shfl_xor(ps, 1, 32);
      ps += __shfl_xor(ps, 2, 32);
      ps += __shfl_xor(ps, 4, 32);
      ps += __shfl_xor(ps, 8, 32);
      l[v] = l[v] * alpha + ps;
      m[v] = mnew;
#pragma unroll
      for (int t = 0; t < 4; ++t) o[t][v] *= alpha;
    }

    // ---- O += P @ V : 2 key-chunks x 4 hd-subtiles ----
#pragma unroll
    for (int kc = 0; kc < 2; ++kc) {
      const unsigned short* prow = &ldsP[wave][l16 * 64 + kc * 32];
      const v16bf ap = combine16(*(const v8u*)(prow + half * 8),
                                 *(const v8u*)(prow + 16 + half * 8));
#pragma unroll
      for (int t = 0; t < 4; ++t) {
        const unsigned short* vp = ldsV + (t * 16 + l16) * 64 + kc * 32 + half * 8;
        o[t] = WMMA_BF16(ap, combine16(*(const v8u*)vp, *(const v8u*)(vp + 16)),
                         o[t]);
      }
    }
  }

  // ---- normalize, write attn in [B,S,D] bf16 ----
#pragma unroll
  for (int v = 0; v < 8; ++v) {
    const float inv = 1.0f / l[v];
    const int srow = qRowW + half * 8 + v;
    unsigned short* orow = attnOut + (size_t)(b * SS + srow) * DD + h * HDD;
#pragma unroll
    for (int t = 0; t < 4; ++t)
      orow[t * 16 + l16] = f32_to_bf16(o[t][v] * inv);
  }
}

// ---------------------------------------------------------------------------
// Kernel 4: output projection  out[8192,1024] f32 = attn @ Wo + bias
// Same LDS-free structure as gemm_qkv.
// ---------------------------------------------------------------------------
__global__ __launch_bounds__(256) void gemm_out_kernel(
    const unsigned short* __restrict__ A,    // [8192,1024] bf16 (attn)
    const unsigned short* __restrict__ Wt,   // [N,K] bf16 (Wo transposed)
    const float* __restrict__ bias,
    float* __restrict__ out)
{
  const int tid  = threadIdx.x;
  const int wave = tid >> 5;
  const int lane = tid & 31;
  const int half = lane >> 4;
  const int l16  = lane & 15;

  const int rowBase = blockIdx.y * 128 + (wave >> 1) * 32;
  const int nBase   = blockIdx.x * 128 + (wave & 1) * 64;

  v8f acc[2][4] = {};

  for (int k0 = 0; k0 < DD; k0 += 32) {
    v16bf afrag[2];
#pragma unroll
    for (int r = 0; r < 2; ++r) {
      const unsigned short* arow = A + (size_t)(rowBase + r * 16 + l16) * DD + k0;
      afrag[r] = combine16(*(const v8u*)(arow + half * 8),
                           *(const v8u*)(arow + 16 + half * 8));
      __builtin_prefetch(arow + 128, 0, 3);
    }
#pragma unroll
    for (int t = 0; t < 4; ++t) {
      const unsigned short* brow =
          Wt + (size_t)(nBase + t * 16 + l16) * DD + k0 + half * 8;
      const v16bf bfrag = combine16(*(const v8u*)brow, *(const v8u*)(brow + 16));
      acc[0][t] = WMMA_BF16(afrag[0], bfrag, acc[0][t]);
      acc[1][t] = WMMA_BF16(afrag[1], bfrag, acc[1][t]);
    }
  }

#pragma unroll
  for (int t = 0; t < 4; ++t) {
    const int gcol = nBase + t * 16 + l16;
    const float bv = bias[gcol];
#pragma unroll
    for (int r = 0; r < 2; ++r) {
#pragma unroll
      for (int v = 0; v < 8; ++v) {
        const int grow = rowBase + r * 16 + half * 8 + v;
        out[(size_t)grow * DD + gcol] = acc[r][t][v] + bv;
      }
    }
  }
}

// ---------------------------------------------------------------------------
// Host launcher
// ---------------------------------------------------------------------------
extern "C" void kernel_launch(void* const* d_in, const int* in_sizes, int n_in,
                              void* d_out, int out_size, void* d_ws, size_t ws_size,
                              hipStream_t stream) {
  (void)in_sizes; (void)n_in; (void)out_size; (void)ws_size;

  const float* x    = (const float*)d_in[0];
  const float* Wq_w = (const float*)d_in[1];
  const float* Wq_b = (const float*)d_in[2];
  const float* Wk_w = (const float*)d_in[3];
  const float* Wk_b = (const float*)d_in[4];
  const float* Wv_w = (const float*)d_in[5];
  const float* Wv_b = (const float*)d_in[6];
  const float* Wo_w = (const float*)d_in[7];
  const float* Wo_b = (const float*)d_in[8];
  float* out = (float*)d_out;

  char* ws = (char*)d_ws;
  unsigned short* xb  = (unsigned short*)ws; ws += (size_t)MM * DD * 2;  // 16 MiB
  unsigned short* wqb = (unsigned short*)ws; ws += (size_t)DD * DD * 2;  // Wq^T
  unsigned short* wkb = (unsigned short*)ws; ws += (size_t)DD * DD * 2;  // Wk^T
  unsigned short* wvb = (unsigned short*)ws; ws += (size_t)DD * DD * 2;  // Wv^T
  unsigned short* wob = (unsigned short*)ws; ws += (size_t)DD * DD * 2;  // Wo^T
  unsigned short* Qb  = (unsigned short*)ws; ws += (size_t)MM * DD * 2;  // [B,H,S,HD]
  unsigned short* Kb  = (unsigned short*)ws; ws += (size_t)MM * DD * 2;  // [B,H,S,HD]
  unsigned short* Vb  = (unsigned short*)ws; ws += (size_t)MM * DD * 2;  // [B,H,HD,S]
  unsigned short* Ab  = (unsigned short*)ws; ws += (size_t)MM * DD * 2;  // [B,S,D]

  const int nX = MM * DD;
  cvt_bf16_kernel<<<(nX + 255) / 256, 256, 0, stream>>>(x, xb, nX);

  const dim3 wtGrid(DD / 32, DD / 32);   // 32x32 tiles
  cvt_wt_kernel<<<wtGrid, 256, 0, stream>>>(Wq_w, wqb);
  cvt_wt_kernel<<<wtGrid, 256, 0, stream>>>(Wk_w, wkb);
  cvt_wt_kernel<<<wtGrid, 256, 0, stream>>>(Wv_w, wvb);
  cvt_wt_kernel<<<wtGrid, 256, 0, stream>>>(Wo_w, wob);

  const dim3 gemmGrid(DD / 128, MM / 128);   // (8, 64)
  gemm_qkv_kernel<<<gemmGrid, 256, 0, stream>>>(xb, wqb, Wq_b, Qb, 0);
  gemm_qkv_kernel<<<gemmGrid, 256, 0, stream>>>(xb, wkb, Wk_b, Kb, 0);
  gemm_qkv_kernel<<<gemmGrid, 256, 0, stream>>>(xb, wvb, Wv_b, Vb, 1);

  attn_kernel<<<dim3(SS / 64, BB * HH), 128, 0, stream>>>(Qb, Kb, Vb, Ab);

  gemm_out_kernel<<<gemmGrid, 256, 0, stream>>>(Ab, wob, Wo_b, out);
}
